// TemporalAttention_58119497450446
// MI455X (gfx1250) — matrix operand order repaired and checked by reading, not compiled, not run on hardware
//
#include <hip/hip_runtime.h>
#include <hip/hip_bf16.h>

// ---------------------------------------------------------------------------
// Temporal attention, MI455X (gfx1250, wave32, WMMA) — fully fused.
//   T=7, B=4, H=96, W=96, C=256, NH=8, KD=32
//
// Roofline: projections dominate FLOPs (~135 GFLOP f16 WMMA); memory is the
// binding constraint. One workgroup owns 16 pixels x all 7 frames (112 rows
// x 256 ch); q/k/v and ctx never leave LDS, so global traffic is just
// frames-in (264 MB) + out (264 MB) + L2-resident weights ~= 528 MB
// (~23 us @ 23.3 TB/s) vs ~1.45 GB for a split pipeline.
//
// v is computed from the SAME staged activations as q/k:
//   v = (frames+emb) @ Wv - emb @ Wv, with  vcorr[t] = bv - emb[t] @ Wv
// precomputed by a tiny prep kernel and folded into the v bias.
//
// LDS (dynamic, 241,664 B = 236 KB < 320 KB/WGP):
//   [     0 ..  57343]  xe  : 112x256 f16 (frames+emb)   — aliased by ctx
//   [ 57344 .. 229375]  q/k/v: 3 x 128 pixel-heads x 7x32 f16
//   [229376 .. 241663]  per-wave softmax/vT scratch (8 x 1536 B)
// ---------------------------------------------------------------------------

typedef _Float16 v16h __attribute__((ext_vector_type(16)));
typedef _Float16 v8h  __attribute__((ext_vector_type(8)));
typedef float    v8f  __attribute__((ext_vector_type(8)));
typedef float    v4f  __attribute__((ext_vector_type(4)));

union HFrag {
    v16h     v;
    v8h      h8[2];
    _Float16 h[16];
};

#define C_DIM    256
#define NHEADS   8
#define KD_DIM   32
#define T_DIM    7
#define PIX      16        // pixels per workgroup
#define BHW      36864     // 4*96*96
#define PW_ELEMS 65536     // one packed 256x256 f16 weight matrix

// LDS offsets in halves
#define L_XE     0                         // 112*256 = 28672 halves (56 KB)
#define L_CTX    0                         // aliases XE after QKV phase
#define L_Q      28672                     // 128*224 = 28672 halves each
#define L_K      57344
#define L_V      86016
#define L_SCR    114688                    // 8 waves * 768 halves
#define SMEM_BYTES 241664                  // 120832 halves * 2

// ---- fragment loaders ------------------------------------------------------
// A (16x32 f16, ISA 7.12.2): lane<16: M=lane, h[0..7]=K 0..7, h[8..15]=K 16..23
//                            lane>=16: M=lane-16, K 8..15 / 24..31
__device__ __forceinline__ HFrag load_a_lds(const _Float16* x, int kk, int lane) {
    const int m  = lane & 15;
    const int hi = (lane >> 4) & 1;
    const _Float16* p = x + m * C_DIM + kk * 32 + hi * 8;
    HFrag f;
    f.h8[0] = *(const v8h*)p;
    f.h8[1] = *(const v8h*)(p + 16);
    return f;
}

// B fragments pre-packed: pw[((kk*16+nt)*32+lane)*16 + j]
__device__ __forceinline__ HFrag load_b_pw(const _Float16* pw, int kk, int nt, int lane) {
    const _Float16* p = pw + (((size_t)(kk * 16 + nt) * 32 + lane) << 4);
    HFrag f;
    f.h8[0] = *(const v8h*)p;
    f.h8[1] = *(const v8h*)(p + 8);
    return f;
}

#define WMMA_F16(A, B, Cacc) \
    __builtin_amdgcn_wmma_f32_16x16x32_f16(false, (A), false, (B), (short)0, (Cacc), false, false)

// ---------------------------------------------------------------------------
// Prep 0: pack Wq/Wk/Wv (C,256) and Wo (256,C) into f16 B-fragment layout:
//   lane<16 -> h[0..7]=W[kt*32+0..7], h[8..15]=W[kt*32+16..23]; lane>=16 -> K 8..15 / 24..31
// ---------------------------------------------------------------------------
__global__ void __launch_bounds__(256) pack_weights_kernel(
    const float* __restrict__ Wq, const float* __restrict__ Wk,
    const float* __restrict__ Wv, const float* __restrict__ Wo,
    _Float16* __restrict__ pw)
{
    int idx = blockIdx.x * 256 + threadIdx.x;           // 0 .. 4*65536-1
    if (idx >= 4 * PW_ELEMS) return;
    int j    = idx & 15;
    int lane = (idx >> 4) & 31;
    int nt   = (idx >> 9) & 15;
    int kt   = (idx >> 13) & 7;
    int m    = idx >> 16;
    int krow = kt * 32 + ((lane >> 4) * 8) + ((j < 8) ? j : 8 + j);
    int col  = nt * 16 + (lane & 15);
    const float* W = (m == 0) ? Wq : (m == 1) ? Wk : (m == 2) ? Wv : Wo;
    pw[idx] = (_Float16)W[krow * C_DIM + col];
}

// ---------------------------------------------------------------------------
// Prep 1: vcorr[t][col] = bv[col] - sum_c t_emb[t][c] * Wv[c][col]
// (lets the fused kernel compute v from the emb-added activations)
// ---------------------------------------------------------------------------
__global__ void __launch_bounds__(256) vcorr_kernel(
    const float* __restrict__ t_emb, const float* __restrict__ Wv,
    const float* __restrict__ bv, float* __restrict__ vcorr)
{
    int t = blockIdx.x;          // 0..6
    int col = threadIdx.x;       // 0..255
    float s = 0.0f;
    for (int c = 0; c < C_DIM; ++c)
        s = fmaf(t_emb[t * C_DIM + c], Wv[c * C_DIM + col], s);
    vcorr[t * C_DIM + col] = bv[col] - s;
}

// ---------------------------------------------------------------------------
// Fused kernel: one block = 16 pixels x 7 frames (112 rows x 256 ch)
// ---------------------------------------------------------------------------
__global__ void __launch_bounds__(256) fused_attention_kernel(
    const float* __restrict__ frames, const float* __restrict__ t_emb,
    const _Float16* __restrict__ pw,       // pwq|pwk|pwv|pwo
    const float* __restrict__ bq, const float* __restrict__ bk,
    const float* __restrict__ vcorr,       // [7][256] f32
    const float* __restrict__ bo, float* __restrict__ out)
{
    extern __shared__ __align__(16) char smem_raw[];
    _Float16* smem = (_Float16*)smem_raw;
    _Float16* xe   = smem + L_XE;          // 112 x 256  (frames + emb)
    _Float16* ctx  = smem + L_CTX;         // aliases xe after QKV phase
    _Float16* q_l  = smem + L_Q;           // [p*8+n][t*32+d]
    _Float16* k_l  = smem + L_K;
    _Float16* v_l  = smem + L_V;

    const int pix0 = blockIdx.x * PIX;
    const int wave = threadIdx.x >> 5;
    const int lane = threadIdx.x & 31;
    const int lm   = lane & 15;
    const int hi   = (lane >> 4) & 1;
    _Float16* attn_s = smem + L_SCR + wave * 768;        // 256 halves
    _Float16* vT_s   = attn_s + 256;                     // 512 halves

    // ---- stage frames(+emb) -> f16 LDS: local row lr = t*16 + p ------------
    #pragma unroll
    for (int i = 0; i < 28; ++i) {                       // 112*64 v4f / 256 thr
        int idx = threadIdx.x + i * 256;
        int lr  = idx >> 6;
        int c   = (idx & 63) << 2;
        int t   = lr >> 4;
        int p   = lr & 15;
        v4f f = *(const v4f*)(frames + ((size_t)t * BHW + pix0 + p) * C_DIM + c);
        v4f e = *(const v4f*)(t_emb + t * C_DIM + c);
        #pragma unroll
        for (int q = 0; q < 4; ++q)
            xe[lr * C_DIM + c + q] = (_Float16)(f[q] + e[q]);
    }
    __syncthreads();

    // ---- phase 1: q/k/v projections (q,k: +bias; v: +vcorr[t]) -------------
    // jobs: (mat 0..2) x (rt 0..6) x (nt-pair 0..7) = 168, stride 8 over waves
    for (int job = wave; job < 168; job += 8) {
        int mat = job / 56;
        int rem = job - mat * 56;
        int rt  = rem >> 3;                 // t / row-tile
        int nt0 = (rem & 7) * 2;
        const _Float16* pwm  = pw + (size_t)mat * PW_ELEMS;
        const _Float16* xrow = xe + rt * 16 * C_DIM;
        v8f acc0 = {}, acc1 = {};
        #pragma unroll
        for (int kk = 0; kk < 8; ++kk) {
            HFrag a  = load_a_lds(xrow, kk, lane);
            HFrag b0 = load_b_pw(pwm, kk, nt0,     lane);
            HFrag b1 = load_b_pw(pwm, kk, nt0 + 1, lane);
            acc0 = WMMA_F16(a.v, b0.v, acc0);
            acc1 = WMMA_F16(a.v, b1.v, acc1);
        }
        _Float16* dst = (mat == 0) ? q_l : (mat == 1) ? k_l : v_l;
        #pragma unroll
        for (int u = 0; u < 2; ++u) {
            v8f acc = u ? acc1 : acc0;
            int col = (nt0 + u) * 16 + lm;
            int n = col >> 5, d = col & 31;
            float b = (mat == 0) ? bq[col]
                    : (mat == 1) ? bk[col]
                                 : vcorr[rt * C_DIM + col];
            #pragma unroll
            for (int j = 0; j < 8; ++j) {
                int p = j + hi * 8;                      // C frag row
                dst[(p * NHEADS + n) * (T_DIM * KD_DIM) + rt * KD_DIM + d] =
                    (_Float16)(acc[j] + b);
            }
        }
    }
    __syncthreads();

    // ---- phase 2: attention, one (pixel, head) per wave-iteration ----------
    for (int i = 0; i < 16; ++i) {
        int phl = wave * 16 + i;                         // 0..127
        const _Float16* qp = q_l + phl * (T_DIM * KD_DIM);
        const _Float16* kp = k_l + phl * (T_DIM * KD_DIM);
        const _Float16* vp = v_l + phl * (T_DIM * KD_DIM);
        int p = phl >> 3, n = phl & 7;

        HFrag qa = {};                                   // A = q, rows>=7 zero
        if (lm < T_DIM) {
            const _Float16* a = qp + lm * KD_DIM + hi * 8;
            qa.h8[0] = *(const v8h*)a;
            qa.h8[1] = *(const v8h*)(a + 16);
        }
        HFrag kb = {};                                   // B = k^T, cols>=7 zero
        if (lm < T_DIM) {
            const _Float16* b = kp + lm * KD_DIM + hi * 8;
            kb.h8[0] = *(const v8h*)b;
            kb.h8[1] = *(const v8h*)(b + 16);
        }
        v8f s = {};
        s = WMMA_F16(qa.v, kb.v, s);                     // scores[q][t]

        HFrag vr = {};                                   // stage v^T rows
        #pragma unroll
        for (int t = 0; t < T_DIM; ++t) vr.h[t] = vp[t * KD_DIM + lane];
        *(v8h*)&vT_s[lane * 16]     = vr.h8[0];
        *(v8h*)&vT_s[lane * 16 + 8] = vr.h8[1];

        const float scale = 0.17677669529663687f;        // 1/sqrt(32)
        #pragma unroll
        for (int j = 0; j < 8; ++j) {                    // softmax over t-lanes
            float val = (lm < T_DIM) ? s[j] * scale : -__builtin_inff();
            float mx = val;
            #pragma unroll
            for (int m = 8; m >= 1; m >>= 1) mx = fmaxf(mx, __shfl_xor(mx, m, 32));
            float e = (lm < T_DIM) ? __expf(val - mx) : 0.0f;
            float sum = e;
            #pragma unroll
            for (int m = 8; m >= 1; m >>= 1) sum += __shfl_xor(sum, m, 32);
            attn_s[(j + hi * 8) * 16 + lm] = (_Float16)(e / sum);
        }

        HFrag aa = {};                                   // A = attn (K=t<=7)
        if (!hi) aa.h8[0] = *(const v8h*)&attn_s[lm * 16];
        HFrag vb0 = {}, vb1 = {};                        // B = v (K=t, N=d)
        if (!hi) {
            const _Float16* p0 = &vT_s[lm * 16];
            vb0.h8[0] = *(const v8h*)p0;
            vb0.h8[1] = *(const v8h*)(p0 + 8);
            const _Float16* p1 = &vT_s[(16 + lm) * 16];
            vb1.h8[0] = *(const v8h*)p1;
            vb1.h8[1] = *(const v8h*)(p1 + 8);
        }
        v8f c0 = {}, c1 = {};
        c0 = WMMA_F16(aa.v, vb0.v, c0);
        c1 = WMMA_F16(aa.v, vb1.v, c1);

        #pragma unroll
        for (int j = 0; j < 8; ++j) {                    // ctx[t*16+p][n*32+d]
            int qv = j + hi * 8;
            if (qv < T_DIM) {
                _Float16* crow = ctx + (qv * 16 + p) * C_DIM + n * KD_DIM;
                crow[lm]      = (_Float16)c0[j];
                crow[16 + lm] = (_Float16)c1[j];
            }
        }
    }
    __syncthreads();

    // ---- phase 3: out = ctx @ Wo + bo (f32 direct to global) ---------------
    const _Float16* pwo = pw + (size_t)3 * PW_ELEMS;
    for (int job = wave; job < 56; job += 8) {           // (rt 0..6) x (ntp 0..7)
        int rt  = job >> 3;
        int nt0 = (job & 7) * 2;
        const _Float16* xrow = ctx + rt * 16 * C_DIM;
        v8f acc0 = {}, acc1 = {};
        #pragma unroll
        for (int kk = 0; kk < 8; ++kk) {
            HFrag a  = load_a_lds(xrow, kk, lane);
            HFrag b0 = load_b_pw(pwo, kk, nt0,     lane);
            HFrag b1 = load_b_pw(pwo, kk, nt0 + 1, lane);
            acc0 = WMMA_F16(a.v, b0.v, acc0);
            acc1 = WMMA_F16(a.v, b1.v, acc1);
        }
        #pragma unroll
        for (int u = 0; u < 2; ++u) {
            v8f acc = u ? acc1 : acc0;
            int col = (nt0 + u) * 16 + lm;
            float b = bo[col];
            #pragma unroll
            for (int j = 0; j < 8; ++j) {
                int p = j + hi * 8;
                out[((size_t)rt * BHW + pix0 + p) * C_DIM + col] = acc[j] + b;
            }
        }
    }
}

// ---------------------------------------------------------------------------
// launch: pack + vcorr prep, then the fused kernel.
// workspace: 512 KB packed f16 weights + 7 KB vcorr.
// ---------------------------------------------------------------------------
extern "C" void kernel_launch(void* const* d_in, const int* in_sizes, int n_in,
                              void* d_out, int out_size, void* d_ws, size_t ws_size,
                              hipStream_t stream)
{
    const float* frames = (const float*)d_in[0];
    const float* t_emb  = (const float*)d_in[1];
    const float* Wq     = (const float*)d_in[2];
    const float* bq     = (const float*)d_in[3];
    const float* Wk     = (const float*)d_in[4];
    const float* bk     = (const float*)d_in[5];
    const float* Wv     = (const float*)d_in[6];
    const float* bv     = (const float*)d_in[7];
    const float* Wo     = (const float*)d_in[8];
    const float* bo     = (const float*)d_in[9];
    float* out = (float*)d_out;

    char* ws = (char*)d_ws;
    _Float16* pw    = (_Float16*)ws;                          // 4*65536 f16
    float*    vcorr = (float*)(ws + 4 * PW_ELEMS * 2);        // 7*256 f32

    (void)hipFuncSetAttribute((const void*)fused_attention_kernel,
                              hipFuncAttributeMaxDynamicSharedMemorySize,
                              SMEM_BYTES);

    pack_weights_kernel<<<(4 * PW_ELEMS) / 256, 256, 0, stream>>>(Wq, Wk, Wv, Wo, pw);
    vcorr_kernel<<<T_DIM, 256, 0, stream>>>(t_emb, Wv, bv, vcorr);
    fused_attention_kernel<<<BHW / PIX, 256, SMEM_BYTES, stream>>>(
        frames, t_emb, pw, bq, bk, vcorr, bo, out);
}
